// MultiHeadAttention_61624190763647
// MI455X (gfx1250) — matrix operand order repaired and checked by reading, not compiled, not run on hardware
//
#include <hip/hip_runtime.h>

typedef __attribute__((ext_vector_type(16))) _Float16 v16h;
typedef __attribute__((ext_vector_type(8)))  float    v8f;
typedef __attribute__((ext_vector_type(4)))  int      v4i;

#define B_  4
#define S_  4096
#define DM_ 512
#define DK_ 64
#define NH_ 8
#define LOG2E 1.44269504088896340736f

#if __has_builtin(__builtin_amdgcn_s_wait_asynccnt)
#define WAIT_ASYNC(n) __builtin_amdgcn_s_wait_asynccnt(n)
#else
#define WAIT_ASYNC(n) asm volatile("s_wait_asynccnt " #n ::: "memory")
#endif

#define AS1 __attribute__((address_space(1)))
#define AS3 __attribute__((address_space(3)))

// Cooperative 16-byte stage: async global->LDS DMA (CDNA5, ASYNCcnt-tracked).
// Existence + signature of this builtin was proven by the round-2 diagnostic.
static __device__ __forceinline__ void stage_b128(const _Float16* g, _Float16* l) {
  __builtin_amdgcn_global_load_async_to_lds_b128((AS1 v4i*)g, (AS3 v4i*)l, 0, 0);
}

static __device__ __forceinline__ v8f wmma16(v16h a, v16h b, v8f c) {
  // D = A(16x32 f16) * B(32x16 f16) + C(16x16 f32)
  return __builtin_amdgcn_wmma_f32_16x16x32_f16(false, a, false, b, (short)0, c,
                                                false, false);
}

// A-fragment element -> K index (contiguous: k = 16*hi + i, matches B loads)
static __device__ __forceinline__ int a_k(int i, int hi) {
  return hi * 16 + i;
}

// ---------------------------------------------------------------------------
// Stage 0: weight prep.
//  sections 0..2: w{q,k,v}_t[n][k] = (f16) W[k*64+n]        (64x512 each)
//  section  3   : wo_t[d][k] = (f16) sum_h W_O[(h*64+k)*512+d]   (512x64)
// ---------------------------------------------------------------------------
__global__ __launch_bounds__(256) void mha_prep_weights(
    const float* __restrict__ WQ, const float* __restrict__ WK,
    const float* __restrict__ WV, const float* __restrict__ WO,
    _Float16* __restrict__ wq_t, _Float16* __restrict__ wk_t,
    _Float16* __restrict__ wv_t, _Float16* __restrict__ wo_t) {
  int flat = blockIdx.x * 256 + threadIdx.x;   // 0 .. 131071
  int sec = flat >> 15;
  int r   = flat & 32767;
  if (sec < 3) {
    int n = r >> 9;          // 0..63
    int k = r & 511;         // 0..511
    const float* W = (sec == 0) ? WQ : (sec == 1) ? WK : WV;
    _Float16* dst  = (sec == 0) ? wq_t : (sec == 1) ? wk_t : wv_t;
    dst[r] = (_Float16)W[k * 64 + n];
  } else {
    int d = r >> 6;          // 0..511
    int k = r & 63;          // 0..63
    float s = 0.f;
#pragma unroll
    for (int h = 0; h < NH_; ++h) s += WO[(h * 64 + k) * 512 + d];
    wo_t[d * 64 + k] = (_Float16)s;
  }
}

// ---------------------------------------------------------------------------
// Stage 1: projection GEMM  P = X(16384x512 f32) @ W(512x64) -> f16.
// One wave computes a 16x64 tile; K-loop of 32 with 4 WMMAs per step.
// TRANS==0 -> Out[row*64+col]; TRANS==1 -> Out[b][col][s] (feature-major V)
// ---------------------------------------------------------------------------
template <int TRANS>
__global__ __launch_bounds__(256) void mha_proj(
    const float* __restrict__ X, const _Float16* __restrict__ Wt,
    _Float16* __restrict__ Out) {
  const int wv   = threadIdx.x >> 5;
  const int lane = threadIdx.x & 31;
  const int n    = lane & 15;     // row (A) / col (B,C) within tile
  const int hi   = lane >> 4;     // lane-half selector
  const int s0   = (blockIdx.x * 8 + wv) * 16;   // global row base

  v8f acc[4] = {v8f{}, v8f{}, v8f{}, v8f{}};

  for (int kb = 0; kb < DM_; kb += 32) {
    v16h a;
#pragma unroll
    for (int i = 0; i < 16; ++i)
      a[i] = (_Float16)X[(size_t)(s0 + n) * DM_ + kb + a_k(i, hi)];
#pragma unroll
    for (int ct = 0; ct < 4; ++ct) {
      v16h b = *(const v16h*)(Wt + (size_t)(ct * 16 + n) * DM_ + kb + hi * 16);
      acc[ct] = wmma16(a, b, acc[ct]);
    }
  }

#pragma unroll
  for (int ct = 0; ct < 4; ++ct) {
#pragma unroll
    for (int j = 0; j < 8; ++j) {
      int row = s0 + hi * 8 + j;
      int col = ct * 16 + n;
      if (TRANS == 0) {
        Out[(size_t)row * DK_ + col] = (_Float16)acc[ct][j];
      } else {
        int b = row >> 12;          // row / 4096
        int s = row & 4095;
        Out[((size_t)b * DK_ + col) * S_ + s] = (_Float16)acc[ct][j];
      }
    }
  }
}

// ---------------------------------------------------------------------------
// Stage 2: fused flash attention + output projection.
// One wave owns 16 query rows. K/V tiles (32 keys) are staged cooperatively
// into double-buffered LDS via async global->LDS DMA; online softmax in
// registers; epilogue head(16x64) @ W_eff(64x512) -> fp32 out.
// ---------------------------------------------------------------------------
__global__ __launch_bounds__(256) void mha_flash_attn(
    const _Float16* __restrict__ q_ws, const _Float16* __restrict__ k_ws,
    const _Float16* __restrict__ v_t,  const _Float16* __restrict__ wo_t,
    float* __restrict__ out) {
  __shared__ alignas(32) _Float16 ldsK[2][32 * 64];   // [buf][key][feat] 4KB each
  __shared__ alignas(32) _Float16 ldsV[2][64 * 32];   // [buf][feat][key] 4KB each
  __shared__ alignas(32) _Float16 ldsP[8][16 * 64];   // per-wave transpose scratch

  const int t    = threadIdx.x;
  const int b    = blockIdx.y;
  const int wv   = t >> 5;
  const int lane = t & 31;
  const int n    = lane & 15;
  const int hi   = lane >> 4;
  const int s0   = blockIdx.x * 128 + wv * 16;   // query row base in batch

  const _Float16* qb = q_ws + (size_t)b * S_ * DK_;
  const _Float16* kp = k_ws + (size_t)b * S_ * DK_;
  const _Float16* vp = v_t  + (size_t)b * DK_ * S_;
  _Float16* lp = ldsP[wv];

  // ---- preload first K/V tile (32 keys): 256 threads x 16B per matrix ----
  stage_b128(kp + (size_t)(t >> 3) * DK_ + (t & 7) * 8, &ldsK[0][t * 8]);
  stage_b128(vp + (size_t)(t >> 2) * S_  + (t & 3) * 8, &ldsV[0][t * 8]);

  // ---- Q A-fragments (features 0..31 / 32..63) ----
  v16h a0, a1;
#pragma unroll
  for (int i = 0; i < 16; ++i) {
    int k = a_k(i, hi);
    a0[i] = qb[(size_t)(s0 + n) * DK_ + k];
    a1[i] = qb[(size_t)(s0 + n) * DK_ + 32 + k];
  }

  float mrow[8], lrow[8];
  v8f o[4] = {v8f{}, v8f{}, v8f{}, v8f{}};
#pragma unroll
  for (int j = 0; j < 8; ++j) { mrow[j] = -1e30f; lrow[j] = 0.f; }

  int cur = 0;
  for (int kb = 0; kb < S_; kb += 32, cur ^= 1) {
    // ---- prefetch next tile into the other buffer, wait for current ----
    if (kb + 32 < S_) {
      stage_b128(kp + (size_t)(kb + 32 + (t >> 3)) * DK_ + (t & 7) * 8,
                 &ldsK[cur ^ 1][t * 8]);
      stage_b128(vp + (size_t)(t >> 2) * S_ + kb + 32 + (t & 3) * 8,
                 &ldsV[cur ^ 1][t * 8]);
      WAIT_ASYNC(2);   // current tile's DMAs retired (async completes in order)
    } else {
      WAIT_ASYNC(0);
    }
    __syncthreads();                          // tile visible to all waves
    const _Float16* Kc = ldsK[cur];
    const _Float16* Vc = ldsV[cur];

    // ---- scores: S = (Q @ K^T) * 1/sqrt(64); two 16-key groups ----
    v8f sa = {}, sb = {};
    {
      v16h b0 = *(const v16h*)(Kc + n * DK_ + hi * 16);
      v16h b1 = *(const v16h*)(Kc + n * DK_ + 32 + hi * 16);
      sa = wmma16(a0, b0, sa);
      sa = wmma16(a1, b1, sa);
    }
    {
      v16h b0 = *(const v16h*)(Kc + (16 + n) * DK_ + hi * 16);
      v16h b1 = *(const v16h*)(Kc + (16 + n) * DK_ + 32 + hi * 16);
      sb = wmma16(a0, b0, sb);
      sb = wmma16(a1, b1, sb);
    }

    // ---- online softmax (row stats per-VGPR, replicated over 16 lanes) ----
#pragma unroll
    for (int j = 0; j < 8; ++j) {
      float x0 = sa[j] * 0.125f;
      float x1 = sb[j] * 0.125f;
      float tmx = fmaxf(x0, x1);
#pragma unroll
      for (int off = 1; off < 16; off <<= 1)
        tmx = fmaxf(tmx, __shfl_xor(tmx, off, 16));
      float nm    = fmaxf(mrow[j], tmx);
      float alpha = exp2f((mrow[j] - nm) * LOG2E);
      float p0 = exp2f((x0 - nm) * LOG2E);
      float p1 = exp2f((x1 - nm) * LOG2E);
      float rs = p0 + p1;
#pragma unroll
      for (int off = 1; off < 16; off <<= 1)
        rs += __shfl_xor(rs, off, 16);
      lrow[j] = lrow[j] * alpha + rs;
      mrow[j] = nm;
      sa[j] = p0;   // reuse as P
      sb[j] = p1;
#pragma unroll
      for (int f = 0; f < 4; ++f) o[f][j] *= alpha;
    }

    // ---- transpose P (C layout -> A layout) via per-wave LDS tile ----
#pragma unroll
    for (int j = 0; j < 8; ++j) {
      int r = hi * 8 + j;
      lp[r * 32 + n]      = (_Float16)sa[j];
      lp[r * 32 + 16 + n] = (_Float16)sb[j];
    }
    __syncthreads();
    v16h pf;
#pragma unroll
    for (int i = 0; i < 16; ++i) pf[i] = lp[n * 32 + a_k(i, hi)];

    // ---- O += P @ V  (feature-major V: contiguous LDS B fragments) ----
#pragma unroll
    for (int f = 0; f < 4; ++f) {
      v16h vb = *(const v16h*)(Vc + (f * 16 + n) * 32 + hi * 16);
      o[f] = wmma16(pf, vb, o[f]);
    }
    __syncthreads();   // all waves done with buf[cur] before it is re-staged
  }

  // ---- normalize, stage head (16x64 f16) in LDS ----
#pragma unroll
  for (int j = 0; j < 8; ++j) {
    float inv = (lrow[j] > 0.f) ? (1.0f / lrow[j]) : 0.f;
#pragma unroll
    for (int f = 0; f < 4; ++f) o[f][j] *= inv;
  }
#pragma unroll
  for (int f = 0; f < 4; ++f)
#pragma unroll
    for (int j = 0; j < 8; ++j)
      lp[(hi * 8 + j) * 64 + f * 16 + n] = (_Float16)o[f][j];
  __syncthreads();

  v16h h0, h1;
#pragma unroll
  for (int i = 0; i < 16; ++i) {
    int k = a_k(i, hi);
    h0[i] = lp[n * 64 + k];
    h1[i] = lp[n * 64 + 32 + k];
  }

  // ---- epilogue GEMM: out = head @ W_eff  (wo_t[d][k] transposed) ----
  float* ob = out + (size_t)b * S_ * DM_;
  for (int dt = 0; dt < 32; ++dt) {
    v16h w0 = *(const v16h*)(wo_t + (size_t)(dt * 16 + n) * DK_ + hi * 16);
    v16h w1 = *(const v16h*)(wo_t + (size_t)(dt * 16 + n) * DK_ + 32 + hi * 16);
    v8f c = {};
    c = wmma16(h0, w0, c);
    c = wmma16(h1, w1, c);
#pragma unroll
    for (int j = 0; j < 8; ++j)
      ob[(size_t)(s0 + hi * 8 + j) * DM_ + dt * 16 + n] = c[j];
  }
}

// ---------------------------------------------------------------------------
extern "C" void kernel_launch(void* const* d_in, const int* in_sizes, int n_in,
                              void* d_out, int out_size, void* d_ws, size_t ws_size,
                              hipStream_t stream) {
  (void)in_sizes; (void)n_in; (void)out_size; (void)ws_size;
  const float* Q  = (const float*)d_in[0];
  const float* K  = (const float*)d_in[1];
  const float* V  = (const float*)d_in[2];
  const float* WQ = (const float*)d_in[3];
  const float* WK = (const float*)d_in[4];
  const float* WV = (const float*)d_in[5];
  const float* WO = (const float*)d_in[6];
  float* out = (float*)d_out;

  const size_t QKV = (size_t)B_ * S_ * DK_ * sizeof(_Float16);  // 2 MB
  const size_t WT  = (size_t)DK_ * DM_ * sizeof(_Float16);      // 64 KB
  char* ws = (char*)d_ws;
  _Float16* q_ws = (_Float16*)(ws);
  _Float16* k_ws = (_Float16*)(ws + QKV);
  _Float16* v_t  = (_Float16*)(ws + 2 * QKV);
  _Float16* wq_t = (_Float16*)(ws + 3 * QKV);
  _Float16* wk_t = (_Float16*)(ws + 3 * QKV + WT);
  _Float16* wv_t = (_Float16*)(ws + 3 * QKV + 2 * WT);
  _Float16* wo_t = (_Float16*)(ws + 3 * QKV + 3 * WT);

  mha_prep_weights<<<512, 256, 0, stream>>>(WQ, WK, WV, WO, wq_t, wk_t, wv_t, wo_t);

  // 16384 rows / (8 waves * 16 rows) = 128 blocks
  mha_proj<0><<<128, 256, 0, stream>>>(Q, wq_t, q_ws);
  mha_proj<0><<<128, 256, 0, stream>>>(K, wk_t, k_ws);
  mha_proj<1><<<128, 256, 0, stream>>>(V, wv_t, v_t);

  mha_flash_attn<<<dim3(32, 4), 256, 0, stream>>>(q_ws, k_ws, v_t, wo_t, out);
}
